// AtomisticModel_49340584296805
// MI455X (gfx1250) — matrix operand order repaired and checked by reading, not compiled, not run on hardware
//
#include <hip/hip_runtime.h>
#include <hip/hip_bf16.h>
#include <math.h>

// ---------------------------------------------------------------------------
// GM-NN atomistic model on gfx1250.
//  - edge moments: f32 atomics into L2-resident 40MB table
//  - gm GEMM:      v_wmma_f32_16x16x4_f32  (wide dynamic range -> fp32)
//  - message + readout GEMMs: v_wmma_f32_16x16x32_f16 (f16 in, f32 accum)
//  - all B matrices pre-packed into per-lane fragment-major layout so the
//    K-loops are pure {global_load + v_wmma} with immediate offsets.
// ---------------------------------------------------------------------------

typedef _Float16 half_t;
typedef __attribute__((ext_vector_type(2)))  float    v2f;
typedef __attribute__((ext_vector_type(8)))  float    v8f;
typedef __attribute__((ext_vector_type(8)))  _Float16 v8h;
typedef __attribute__((ext_vector_type(16))) _Float16 v16h;

#define N_ATOMS     50000
#define N_ATOMS_PAD 50016            // 3126 M-tiles of 16
#define N_EDGES     800000           // 50000 M-tiles of 16
#define NODE_DIM    64
#define GM_DIM      330
#define GM_PAD      336              // K padded to multiple of 4
#define HID         512
#define R_MAXF      6.0f

// ---- workspace layout (byte offsets, 256B aligned regions) ----------------
static constexpr size_t al256(size_t x) { return (x + 255) & ~(size_t)255; }
static constexpr size_t B_M   = 0;                                            // f32 [50000,200]
static constexpr size_t B_GM  = al256(B_M   + (size_t)N_ATOMS    * 200 * 4);  // f32 [50016,336]
static constexpr size_t B_H32 = al256(B_GM  + (size_t)N_ATOMS_PAD * GM_PAD * 4); // f32 [50016,64]
static constexpr size_t B_U   = al256(B_H32 + (size_t)N_ATOMS_PAD * 64 * 4);  // f32 [50016,64]
static constexpr size_t B_H16 = al256(B_U   + (size_t)N_ATOMS_PAD * 64 * 4);  // f16 [50016,64]
static constexpr size_t B_H2  = al256(B_H16 + (size_t)N_ATOMS_PAD * 64 * 2);  // f16 [50016,64]
static constexpr size_t B_A1  = al256(B_H2  + (size_t)N_ATOMS_PAD * 64 * 2);  // f16 [50016,512]
static constexpr size_t B_A2  = al256(B_A1  + (size_t)N_ATOMS_PAD * HID * 2); // f16 [50016,512]
static constexpr size_t B_WNP = al256(B_A2  + (size_t)N_ATOMS_PAD * HID * 2); // f32 packed [336,64]
static constexpr size_t B_WMP = al256(B_WNP + (size_t)GM_PAD * 64 * 4);       // f16 packed [64,64]
static constexpr size_t B_W1P = al256(B_WMP + (size_t)64 * 64 * 2);           // f16 packed [64,512]
static constexpr size_t B_W2P = al256(B_W1P + (size_t)64 * HID * 2);          // f16 packed [512,512]

__device__ __forceinline__ float swish_f(float x) { return x / (1.0f + __expf(-x)); }
__device__ __forceinline__ v16h cat8(v8h lo, v8h hi) {
    return __builtin_shufflevector(lo, hi, 0,1,2,3,4,5,6,7,8,9,10,11,12,13,14,15);
}

// ---------------------------------------------------------------------------
// K1: per-edge radial/direction moments, atomically accumulated into
//     M[n,200] = {m0[5], m1[5*3], m2[5*9], m3[5*27]}
// ---------------------------------------------------------------------------
__global__ void k_edge_moments(const float* __restrict__ dr,
                               const int*   __restrict__ iidx,
                               float*       __restrict__ M)
{
    int e = blockIdx.x * blockDim.x + threadIdx.x;
    if (e >= N_EDGES) return;
    float dx = dr[3 * e + 0], dy = dr[3 * e + 1], dz = dr[3 * e + 2];
    float r   = sqrtf(dx * dx + dy * dy + dz * dz);
    float inv = 1.0f / (r + 1e-9f);
    float uu[3] = { dx * inv, dy * inv, dz * inv };
    float cut = (r < R_MAXF) ? 0.5f * (__cosf(3.14159265358979323846f * r / R_MAXF) + 1.0f) : 0.0f;
    if (cut == 0.0f) return;
    float* row = M + (size_t)iidx[e] * 200;
    #pragma unroll
    for (int k = 0; k < 5; k++) {
        float d = r - (0.5f + 1.375f * (float)k);       // linspace(0.5, 6.0, 5)
        float g = __expf(-d * d) * cut;
        atomicAdd(row + k, g);
        float* p1 = row + 5  + k * 3;
        float* p2 = row + 20 + k * 9;
        float* p3 = row + 65 + k * 27;
        #pragma unroll
        for (int a = 0; a < 3; a++) {
            float ga = g * uu[a];
            atomicAdd(p1 + a, ga);
            #pragma unroll
            for (int b = 0; b < 3; b++) {
                float gab = ga * uu[b];
                atomicAdd(p2 + a * 3 + b, gab);
                #pragma unroll
                for (int c = 0; c < 3; c++)
                    atomicAdd(p3 + a * 9 + b * 3 + c, gab * uu[c]);
            }
        }
    }
}

// ---------------------------------------------------------------------------
// K2: per-atom gaussian-moment invariants -> gm[n, 336] (K-padded)
// ---------------------------------------------------------------------------
__global__ void k_contract(const float* __restrict__ M, float* __restrict__ gm)
{
    int n = blockIdx.x * blockDim.x + threadIdx.x;
    if (n >= N_ATOMS) return;
    const float* mm = M + (size_t)n * 200;
    float*       o  = gm + (size_t)n * GM_PAD;

    float m1[15], m2[45];
    #pragma unroll
    for (int t = 0; t < 15; t++) m1[t] = mm[5 + t];
    #pragma unroll
    for (int t = 0; t < 45; t++) m2[t] = mm[20 + t];
    const float* m3 = mm + 65;

    #pragma unroll
    for (int k = 0; k < 5; k++) o[k] = mm[k];
    for (int r = 0; r < 5; r++)
        for (int s = 0; s < 5; s++) {
            float acc = 0.0f;
            #pragma unroll
            for (int a = 0; a < 3; a++) acc += m1[r * 3 + a] * m1[s * 3 + a];
            o[5 + r * 5 + s] = acc;
        }
    for (int r = 0; r < 5; r++)
        for (int s = 0; s < 5; s++) {
            float acc = 0.0f;
            #pragma unroll
            for (int e = 0; e < 9; e++) acc += m2[r * 9 + e] * m2[s * 9 + e];
            o[30 + r * 5 + s] = acc;
        }
    for (int r = 0; r < 5; r++) {
        float a3[27];
        #pragma unroll
        for (int e = 0; e < 27; e++) a3[e] = m3[r * 27 + e];
        for (int s = 0; s < 5; s++) {
            float acc = 0.0f;
            #pragma unroll
            for (int e = 0; e < 27; e++) acc += a3[e] * m3[s * 27 + e];
            o[55 + r * 5 + s] = acc;
        }
    }
    for (int r = 0; r < 5; r++)
        for (int s = 0; s < 5; s++)
            for (int t = 0; t < 5; t++) {
                float acc = 0.0f;
                #pragma unroll
                for (int i = 0; i < 3; i++)
                    #pragma unroll
                    for (int j = 0; j < 3; j++)
                        acc += m2[r * 9 + i * 3 + j] * m1[s * 3 + i] * m1[t * 3 + j];
                o[80 + (r * 5 + s) * 5 + t] = acc;
            }
    for (int r = 0; r < 5; r++) {
        float a3[27];
        #pragma unroll
        for (int e = 0; e < 27; e++) a3[e] = m3[r * 27 + e];
        for (int s = 0; s < 5; s++)
            for (int t = 0; t < 5; t++) {
                float acc = 0.0f;
                #pragma unroll
                for (int i = 0; i < 3; i++)
                    #pragma unroll
                    for (int j = 0; j < 3; j++) {
                        float m2v = m2[s * 9 + i * 3 + j];
                        #pragma unroll
                        for (int k = 0; k < 3; k++)
                            acc += a3[i * 9 + j * 3 + k] * m2v * m1[t * 3 + k];
                    }
                o[205 + (r * 5 + s) * 5 + t] = acc;
            }
    }
    #pragma unroll
    for (int k = GM_DIM; k < GM_PAD; k++) o[k] = 0.0f;
}

// ---------------------------------------------------------------------------
// B-packers: rearrange row-major B[K,N] (f32) into per-lane fragment-major
// layouts so the GEMM K-loops load one contiguous chunk per tile.
// f32 frag (16x16x4): lane l holds B[k + (l>>4)*2 + {0,1}][tile*16 + (l&15)]
// f16 frag (16x16x32): lane l holds B[kb + (l>>4)*8 + {0..7, 16..23}][tile*16 + (l&15)]
// ---------------------------------------------------------------------------
__global__ void k_pack_b32(const float* __restrict__ B, float* __restrict__ Bp,
                           int Ksrc, int Kpad, int N)
{
    int i = blockIdx.x * blockDim.x + threadIdx.x;
    int total = (Kpad / 4) * (N / 16) * 32;
    if (i >= total) return;
    int lane  = i & 31;
    int tile  = (i >> 5) % (N / 16);
    int kstep = (i >> 5) / (N / 16);
    int col = tile * 16 + (lane & 15);
    int k   = kstep * 4 + (lane >> 4) * 2;
    float* p = Bp + (size_t)i * 2;
    p[0] = (k     < Ksrc) ? B[(size_t)k       * N + col] : 0.0f;
    p[1] = (k + 1 < Ksrc) ? B[(size_t)(k + 1) * N + col] : 0.0f;
}

__global__ void k_pack_b16(const float* __restrict__ B, half_t* __restrict__ Bp,
                           int K, int N)
{
    int i = blockIdx.x * blockDim.x + threadIdx.x;
    int total = (K / 32) * (N / 16) * 32;
    if (i >= total) return;
    int lane  = i & 31;
    int tile  = (i >> 5) % (N / 16);
    int kstep = (i >> 5) / (N / 16);
    int col   = tile * 16 + (lane & 15);
    int kbase = kstep * 32 + (lane >> 4) * 8;
    half_t* p = Bp + (size_t)i * 16;
    #pragma unroll
    for (int q = 0; q < 16; q++) p[q] = (half_t)B[(size_t)(kbase + q + (q >> 3) * 8) * N + col];
}

// ---------------------------------------------------------------------------
// fp32 GEMM (v_wmma_f32_16x16x4_f32): C = A[M,K] @ B / ntk-scale.
// Writes both f32 and f16 copies of the result (f16 feeds the message GEMM).
// grid = (M/16, NT/4), block = 32. K, NT compile-time -> fully unrolled.
// ---------------------------------------------------------------------------
template<int K, int NT>
__global__ void k_gemm_f32(const float* __restrict__ A, int lda,
                           const float* __restrict__ Bp,
                           float* __restrict__ C32, half_t* __restrict__ C16,
                           float ntk)
{
    constexpr int N = NT * 16;
    const int lane = threadIdx.x;
    const int m16  = lane & 15;
    const int kh   = lane >> 4;
    const int rowb = blockIdx.x << 4;
    const int t0   = blockIdx.y << 2;
    const int col0 = (t0 << 4) + m16;

    const float* arow = A + (size_t)(rowb + m16) * lda + kh * 2;
    const float* bf   = Bp + ((size_t)t0 * 32 + lane) * 2;

    v8f acc0 = {}, acc1 = {}, acc2 = {}, acc3 = {};
    #pragma unroll
    for (int k = 0; k < K; k += 4) {
        v2f a = *(const v2f*)(arow + k);
        const float* bk = bf + (size_t)(k >> 2) * (NT * 64);
        v2f b0 = *(const v2f*)(bk + 0);
        v2f b1 = *(const v2f*)(bk + 64);
        v2f b2 = *(const v2f*)(bk + 128);
        v2f b3 = *(const v2f*)(bk + 192);
        acc0 = __builtin_amdgcn_wmma_f32_16x16x4_f32(false, a, false, b0, (short)0, acc0, false, false);
        acc1 = __builtin_amdgcn_wmma_f32_16x16x4_f32(false, a, false, b1, (short)0, acc1, false, false);
        acc2 = __builtin_amdgcn_wmma_f32_16x16x4_f32(false, a, false, b2, (short)0, acc2, false, false);
        acc3 = __builtin_amdgcn_wmma_f32_16x16x4_f32(false, a, false, b3, (short)0, acc3, false, false);
    }
    #pragma unroll
    for (int d = 0; d < 8; d++) {
        size_t row = (size_t)(rowb + d + kh * 8);
        float v0 = acc0[d] * ntk, v1 = acc1[d] * ntk;
        float v2 = acc2[d] * ntk, v3 = acc3[d] * ntk;
        float*  c32 = C32 + row * N + col0;
        half_t* c16 = C16 + row * N + col0;
        c32[0] = v0; c32[16] = v1; c32[32] = v2; c32[48] = v3;
        c16[0] = (half_t)v0; c16[16] = (half_t)v1; c16[32] = (half_t)v2; c16[48] = (half_t)v3;
    }
}

// ---------------------------------------------------------------------------
// fp16 GEMM (v_wmma_f32_16x16x32_f16, f32 accumulate):
//   C_f16 = swish( A[M,K]_f16 @ B * ntk + bias )
// grid = (M/16, NT/4), block = 32. K, NT compile-time -> fully unrolled.
// ---------------------------------------------------------------------------
template<int K, int NT>
__global__ void k_gemm_f16(const half_t* __restrict__ A, int lda,
                           const half_t* __restrict__ Bp,
                           half_t* __restrict__ C,
                           float ntk, const float* __restrict__ bias)
{
    constexpr int N = NT * 16;
    const int lane = threadIdx.x;
    const int m16  = lane & 15;
    const int kh   = lane >> 4;
    const int rowb = blockIdx.x << 4;
    const int t0   = blockIdx.y << 2;
    const int col0 = (t0 << 4) + m16;

    const half_t* arow = A + (size_t)(rowb + m16) * lda + kh * 8;
    const half_t* bf   = Bp + ((size_t)t0 * 32 + lane) * 16;

    v8f acc0 = {}, acc1 = {}, acc2 = {}, acc3 = {};
    #pragma unroll
    for (int kb = 0; kb < K; kb += 32) {
        v16h a = cat8(*(const v8h*)(arow + kb), *(const v8h*)(arow + kb + 16));
        const half_t* bk = bf + (size_t)(kb >> 5) * (NT * 512);
        v16h b0 = cat8(*(const v8h*)(bk + 0),    *(const v8h*)(bk + 8));
        v16h b1 = cat8(*(const v8h*)(bk + 512),  *(const v8h*)(bk + 520));
        v16h b2 = cat8(*(const v8h*)(bk + 1024), *(const v8h*)(bk + 1032));
        v16h b3 = cat8(*(const v8h*)(bk + 1536), *(const v8h*)(bk + 1544));
        acc0 = __builtin_amdgcn_wmma_f32_16x16x32_f16(false, a, false, b0, (short)0, acc0, false, false);
        acc1 = __builtin_amdgcn_wmma_f32_16x16x32_f16(false, a, false, b1, (short)0, acc1, false, false);
        acc2 = __builtin_amdgcn_wmma_f32_16x16x32_f16(false, a, false, b2, (short)0, acc2, false, false);
        acc3 = __builtin_amdgcn_wmma_f32_16x16x32_f16(false, a, false, b3, (short)0, acc3, false, false);
    }
    float bs0 = bias[col0], bs1 = bias[col0 + 16], bs2 = bias[col0 + 32], bs3 = bias[col0 + 48];
    #pragma unroll
    for (int d = 0; d < 8; d++) {
        half_t* crow = C + (size_t)(rowb + d + kh * 8) * N + col0;
        crow[0]  = (half_t)swish_f(acc0[d] * ntk + bs0);
        crow[16] = (half_t)swish_f(acc1[d] * ntk + bs1);
        crow[32] = (half_t)swish_f(acc2[d] * ntk + bs2);
        crow[48] = (half_t)swish_f(acc3[d] * ntk + bs3);
    }
}

// ---------------------------------------------------------------------------
// message: u[j] += swish( (h[j] @ W_msg) / 8 ).  f16 gather-GEMM, f32 scatter.
// grid = (N_EDGES/16), block = 32.
// ---------------------------------------------------------------------------
__global__ void k_edge_msg(const half_t* __restrict__ hh,
                           const int*    __restrict__ jidx,
                           const half_t* __restrict__ Bp,
                           float*        __restrict__ u)
{
    const int lane = threadIdx.x;
    const int m16  = lane & 15;
    const int kh   = lane >> 4;
    const int rowb = blockIdx.x << 4;

    const int jg = jidx[rowb + m16];
    const half_t* arow = hh + (size_t)jg * NODE_DIM + kh * 8;
    const half_t* bf   = Bp + (size_t)lane * 16;          // NT = 4, t0 = 0

    v8f acc0 = {}, acc1 = {}, acc2 = {}, acc3 = {};
    #pragma unroll
    for (int kb = 0; kb < NODE_DIM; kb += 32) {
        v16h a = cat8(*(const v8h*)(arow + kb), *(const v8h*)(arow + kb + 16));
        const half_t* bk = bf + (size_t)(kb >> 5) * 2048;
        v16h b0 = cat8(*(const v8h*)(bk + 0),    *(const v8h*)(bk + 8));
        v16h b1 = cat8(*(const v8h*)(bk + 512),  *(const v8h*)(bk + 520));
        v16h b2 = cat8(*(const v8h*)(bk + 1024), *(const v8h*)(bk + 1032));
        v16h b3 = cat8(*(const v8h*)(bk + 1536), *(const v8h*)(bk + 1544));
        acc0 = __builtin_amdgcn_wmma_f32_16x16x32_f16(false, a, false, b0, (short)0, acc0, false, false);
        acc1 = __builtin_amdgcn_wmma_f32_16x16x32_f16(false, a, false, b1, (short)0, acc1, false, false);
        acc2 = __builtin_amdgcn_wmma_f32_16x16x32_f16(false, a, false, b2, (short)0, acc2, false, false);
        acc3 = __builtin_amdgcn_wmma_f32_16x16x32_f16(false, a, false, b3, (short)0, acc3, false, false);
    }
    const float s = 0.125f;                               // 1/sqrt(64)
    #pragma unroll
    for (int d = 0; d < 8; d++) {
        int ja = jidx[rowb + d + kh * 8];
        float* urow = u + (size_t)ja * NODE_DIM + m16;
        atomicAdd(urow + 0,  swish_f(acc0[d] * s));
        atomicAdd(urow + 16, swish_f(acc1[d] * s));
        atomicAdd(urow + 32, swish_f(acc2[d] * s));
        atomicAdd(urow + 48, swish_f(acc3[d] * s));
    }
}

// ---------------------------------------------------------------------------
// residual: h2_f16 = (f16)(h_f32 + u_f32)
// ---------------------------------------------------------------------------
__global__ void k_residual(const float* __restrict__ h, const float* __restrict__ u,
                           half_t* __restrict__ h2)
{
    int i = blockIdx.x * blockDim.x + threadIdx.x;
    if (i >= N_ATOMS_PAD * NODE_DIM) return;
    h2[i] = (half_t)(h[i] + u[i]);
}

// ---------------------------------------------------------------------------
// readout: out = mask * ((a2 . W_r3)/sqrt(512) + b3) * scale[Z] + shift[Z]
// one wave per atom, butterfly reduction.
// ---------------------------------------------------------------------------
__global__ void k_readout(const half_t* __restrict__ a2, const float* __restrict__ Wr3,
                          const float* __restrict__ b3, const int* __restrict__ Z,
                          const float* __restrict__ scale, const float* __restrict__ shift,
                          float* __restrict__ out)
{
    int gtid = blockIdx.x * blockDim.x + threadIdx.x;
    int wave = gtid >> 5;
    int lane = gtid & 31;
    if (wave >= N_ATOMS) return;
    const half_t* row = a2 + (size_t)wave * HID;
    float s = 0.0f;
    #pragma unroll
    for (int k = lane; k < HID; k += 32) s += (float)row[k] * Wr3[k];
    #pragma unroll
    for (int off = 16; off > 0; off >>= 1) s += __shfl_xor(s, off, 32);
    if (lane == 0) {
        float v = s * 0.04419417382415922f + b3[0];       // 1/sqrt(512)
        int z = Z[wave];
        v = v * scale[z] + shift[z];
        out[wave] = (z != 0) ? v : 0.0f;
    }
}

// ---------------------------------------------------------------------------
extern "C" void kernel_launch(void* const* d_in, const int* in_sizes, int n_in,
                              void* d_out, int out_size, void* d_ws, size_t ws_size,
                              hipStream_t stream)
{
    (void)in_sizes; (void)n_in; (void)out_size; (void)ws_size;
    const float* dr    = (const float*)d_in[0];
    const int*   Z     = (const int*)  d_in[1];
    const int*   idx   = (const int*)  d_in[2];   // [2, N_EDGES]
    const float* Wnode = (const float*)d_in[3];
    const float* Wmsg  = (const float*)d_in[4];
    const float* Wr1   = (const float*)d_in[5];
    const float* br1   = (const float*)d_in[6];
    const float* Wr2   = (const float*)d_in[7];
    const float* br2   = (const float*)d_in[8];
    const float* Wr3   = (const float*)d_in[9];
    const float* br3   = (const float*)d_in[10];
    const float* scl   = (const float*)d_in[11];
    const float* shf   = (const float*)d_in[12];
    float* out = (float*)d_out;

    char* ws = (char*)d_ws;
    float*  Mm  = (float*) (ws + B_M);
    float*  gm  = (float*) (ws + B_GM);
    float*  h32 = (float*) (ws + B_H32);
    float*  u   = (float*) (ws + B_U);
    half_t* h16 = (half_t*)(ws + B_H16);
    half_t* h2  = (half_t*)(ws + B_H2);
    half_t* a1  = (half_t*)(ws + B_A1);
    half_t* a2  = (half_t*)(ws + B_A2);
    float*  WnP = (float*) (ws + B_WNP);
    half_t* WmP = (half_t*)(ws + B_WMP);
    half_t* W1P = (half_t*)(ws + B_W1P);
    half_t* W2P = (half_t*)(ws + B_W2P);

    // zero accumulators + gm padding rows (every call; ws is poisoned once)
    hipMemsetAsync(Mm, 0, (size_t)N_ATOMS * 200 * 4, stream);
    hipMemsetAsync(u,  0, (size_t)N_ATOMS_PAD * 64 * 4, stream);
    hipMemsetAsync(gm + (size_t)N_ATOMS * GM_PAD, 0,
                   (size_t)(N_ATOMS_PAD - N_ATOMS) * GM_PAD * 4, stream);

    // pack weights into fragment-major layouts
    k_pack_b32<<<(84 * 4 * 32) / 256, 256, 0, stream>>>(Wnode, WnP, GM_DIM, GM_PAD, NODE_DIM);
    k_pack_b16<<<1, 256, 0, stream>>>(Wmsg, WmP, NODE_DIM, NODE_DIM);          // 256 frags
    k_pack_b16<<<8, 256, 0, stream>>>(Wr1, W1P, NODE_DIM, HID);                // 2048
    k_pack_b16<<<64, 256, 0, stream>>>(Wr2, W2P, HID, HID);                    // 16384

    // descriptor
    k_edge_moments<<<N_EDGES / 256, 256, 0, stream>>>(dr, idx, Mm);
    k_contract<<<(N_ATOMS + 255) / 256, 256, 0, stream>>>(Mm, gm);

    // h = gm @ W_node / sqrt(330)   (fp32 WMMA; also emits f16 copy)
    k_gemm_f32<GM_PAD, 4><<<dim3(N_ATOMS_PAD / 16, 1), 32, 0, stream>>>(
        gm, GM_PAD, WnP, h32, h16, 0.05504818825631803f);

    // u[j] += swish(h[j] @ W_msg / 8)   (f16 WMMA gather-GEMM)
    k_edge_msg<<<dim3(N_EDGES / 16), 32, 0, stream>>>(h16, idx + N_EDGES, WmP, u);

    // h2 = h + u  (f16)
    k_residual<<<(N_ATOMS_PAD * 64) / 256, 256, 0, stream>>>(h32, u, h2);

    // a1 = swish(h2 @ W_r1 / 8 + b1)
    k_gemm_f16<NODE_DIM, 32><<<dim3(N_ATOMS_PAD / 16, 8), 32, 0, stream>>>(
        h2, NODE_DIM, W1P, a1, 0.125f, br1);

    // a2 = swish(a1 @ W_r2 / sqrt(512) + b2)
    k_gemm_f16<HID, 32><<<dim3(N_ATOMS_PAD / 16, 8), 32, 0, stream>>>(
        a1, HID, W2P, a2, 0.04419417382415922f, br2);

    // out = mask * ((a2 . W_r3)/sqrt(512) + b3) * scale[Z] + shift[Z]
    k_readout<<<(N_ATOMS * 32) / 256, 256, 0, stream>>>(a2, Wr3, br3, Z, scl, shf, out);
}
